// MultiHeadGATLayer_65068754534725
// MI455X (gfx1250) — matrix operand order repaired and checked by reading, not compiled, not run on hardware
//
#include <hip/hip_runtime.h>
#include <math.h>

#define N_NODES 50000
#define E_EDGES 800000
#define ET_EDGES (E_EDGES + N_NODES)   // 850000 (with self loops)
#define HID 256
#define NHEAD 8
#define CHD 32
#define FFN_DIM 1024
#define HSTR 256    // fp32 h-tile LDS row stride (contiguous; TDM-friendly)
#define AHSTR 264   // bf16 A-fragment LDS row stride (padded: rows 4 banks apart)
#define GSTR 136    // bf16 g-fragment LDS row stride (padded)

typedef __attribute__((ext_vector_type(16))) __bf16 v16bf;
typedef __attribute__((ext_vector_type(8)))  __bf16 v8bf;
typedef __attribute__((ext_vector_type(8)))  float  v8f;
typedef __attribute__((ext_vector_type(4)))  unsigned u32x4;
typedef __attribute__((ext_vector_type(8)))  int      i32x8;
typedef __attribute__((ext_vector_type(4)))  int      i32x4;

// ---------------------------------------------------------------------------
// bf16 hi/lo split helpers (error-compensated fp32-via-bf16 WMMA)
// ---------------------------------------------------------------------------
__device__ __forceinline__ void split8(const float* f, v8bf& h, v8bf& l) {
#pragma unroll
  for (int j = 0; j < 8; ++j) {
    __bf16 hh = (__bf16)f[j];
    h[j] = hh;
    l[j] = (__bf16)(f[j] - (float)hh);
  }
}

// A fragment, 16-bit A-matrix 16x32 layout (contiguous, matches B):
// lane<16: elems 0..15 = K[0..15]
// lane>=16: elems 0..15 = K[16..31]
__device__ __forceinline__ v16bf load_a16(const __bf16* __restrict__ p, int kb) {
  v8bf a = *(const v8bf*)(p + kb);
  v8bf b = *(const v8bf*)(p + kb + 8);
  return __builtin_shufflevector(a, b, 0, 1, 2, 3, 4, 5, 6, 7,
                                 8, 9, 10, 11, 12, 13, 14, 15);
}

// B fragment, 16-bit B-matrix 32x16 layout (contiguous 16 K per lane)
__device__ __forceinline__ v16bf load_b16(const __bf16* __restrict__ p, int kb) {
  return *(const v16bf*)(p + kb);
}

__device__ __forceinline__ v8f wmma3(v16bf ah, v16bf al, v16bf bh, v16bf bl, v8f c) {
  c = __builtin_amdgcn_wmma_f32_16x16x32_bf16(false, ah, false, bh, (short)0, c, false, false);
  c = __builtin_amdgcn_wmma_f32_16x16x32_bf16(false, ah, false, bl, (short)0, c, false, false);
  c = __builtin_amdgcn_wmma_f32_16x16x32_bf16(false, al, false, bh, (short)0, c, false, false);
  return c;
}

__device__ __forceinline__ float wave_sum(float s) {
#pragma unroll
  for (int off = 16; off > 0; off >>= 1) s += __shfl_xor(s, off, 32);
  return s;
}

// ---------------------------------------------------------------------------
// Kernel 0: one-time fp32 -> bf16 hi/lo split of a weight matrix
// ---------------------------------------------------------------------------
__global__ __launch_bounds__(256) void cvt_weights_kernel(const float* __restrict__ w,
                                                          __bf16* __restrict__ hi,
                                                          __bf16* __restrict__ lo, int n) {
  const int i = blockIdx.x * 256 + threadIdx.x;
  if (i < n) {
    float f = w[i];
    __bf16 h = (__bf16)f;
    hi[i] = h;
    lo[i] = (__bf16)(f - (float)h);
  }
}

// ---------------------------------------------------------------------------
// Kernel 1: C[M,Nout] = A[M,256] @ W[Nout,256]^T, W pre-split to bf16 hi/lo.
// grid = (M/16, Nout/128), block = 256 (8 waves, one 16x16 tile per wave)
// ---------------------------------------------------------------------------
__global__ __launch_bounds__(256) void gemm16_k256(const float* __restrict__ A,
                                                   const __bf16* __restrict__ Whi,
                                                   const __bf16* __restrict__ Wlo,
                                                   float* __restrict__ Cout, int Nout) {
  __shared__ __bf16 lds_ah[16 * AHSTR], lds_al[16 * AHSTR];
  const int tid = threadIdx.x;
  const int m0 = blockIdx.x * 16;
  const float* Atile = A + (size_t)m0 * 256;
  {  // stage + split A tile: each thread converts 16 consecutive floats
    const int row = tid >> 4, colb = (tid & 15) * 16;
    const float4* s4 = (const float4*)(Atile + row * 256 + colb);
    float f[16];
    float4 q;
#pragma unroll
    for (int i = 0; i < 4; ++i) {
      q = s4[i];
      f[4 * i + 0] = q.x; f[4 * i + 1] = q.y; f[4 * i + 2] = q.z; f[4 * i + 3] = q.w;
    }
    v8bf h0, h1, l0, l1;
    split8(f, h0, l0); split8(f + 8, h1, l1);
    *(v8bf*)(lds_ah + row * AHSTR + colb) = h0;
    *(v8bf*)(lds_ah + row * AHSTR + colb + 8) = h1;
    *(v8bf*)(lds_al + row * AHSTR + colb) = l0;
    *(v8bf*)(lds_al + row * AHSTR + colb + 8) = l1;
  }
  __syncthreads();
  const int w = tid >> 5, lane = tid & 31, nl = lane & 15;
  const int n = blockIdx.y * 128 + w * 16 + nl;
  const __bf16* whrow = Whi + (size_t)n * 256;
  const __bf16* wlrow = Wlo + (size_t)n * 256;
  const __bf16* ahrow = lds_ah + nl * AHSTR;
  const __bf16* alrow = lds_al + nl * AHSTR;
  const int kba = (lane & 16) ? 16 : 0;
  const int kbb = (lane & 16) ? 16 : 0;
  v8f acc = {0.f, 0.f, 0.f, 0.f, 0.f, 0.f, 0.f, 0.f};
  for (int k0 = 0; k0 < 256; k0 += 32) {
    v16bf ah = load_a16(ahrow, k0 + kba);
    v16bf al = load_a16(alrow, k0 + kba);
    v16bf bh = load_b16(whrow, k0 + kbb);
    v16bf bl = load_b16(wlrow, k0 + kbb);
    acc = wmma3(ah, al, bh, bl, acc);
  }
  const int rbase = m0 + ((lane & 16) ? 8 : 0);
#pragma unroll
  for (int r = 0; r < 8; ++r)
    Cout[(size_t)(rbase + r) * Nout + n] = acc[r];
}

// ---------------------------------------------------------------------------
// Kernel 2: init aggregator (=d_out), segment-max keys, denominators
// ---------------------------------------------------------------------------
__global__ __launch_bounds__(256) void init_kernel(float* __restrict__ agg,
                                                   unsigned* __restrict__ menc,
                                                   float* __restrict__ denom) {
  const int i = blockIdx.x * 256 + threadIdx.x;
  if (i < N_NODES * HID) agg[i] = 0.f;
  if (i < N_NODES * NHEAD) { menc[i] = 0x007FFFFFu; denom[i] = 0.f; }  // enc(-inf)
}

// monotonic float<->uint ordering encode
__device__ __forceinline__ unsigned enc_f(float f) {
  unsigned u = __float_as_uint(f);
  return ((int)u >= 0) ? (u | 0x80000000u) : ~u;
}
__device__ __forceinline__ float dec_f(unsigned e) {
  unsigned u = (e & 0x80000000u) ? (e & 0x7FFFFFFFu) : ~e;
  return __uint_as_float(u);
}

// ---------------------------------------------------------------------------
// Kernel 3: per-(edge,head) attention logit + leaky relu + segment max
// ---------------------------------------------------------------------------
__global__ __launch_bounds__(256) void edge_alpha_kernel(
    const int* __restrict__ src, const int* __restrict__ dst,
    const int* __restrict__ etype,
    const float* __restrict__ xsrc, const float* __restrict__ xdst,
    const float* __restrict__ att_s, const float* __restrict__ att_d,
    float* __restrict__ alpha_out, unsigned* __restrict__ menc) {
  __shared__ float s_as[2 * NHEAD * CHD], s_ad[2 * NHEAD * CHD];
  for (int i = threadIdx.x; i < 2 * NHEAD * CHD; i += 256) { s_as[i] = att_s[i]; s_ad[i] = att_d[i]; }
  __syncthreads();
  const int gid = blockIdx.x * 256 + threadIdx.x;
  const int e = gid >> 3, h = gid & 7;
  if (e >= ET_EDGES) return;
  int s, d, t;
  if (e < E_EDGES) { s = src[e]; d = dst[e]; t = etype[e]; }
  else             { s = e - E_EDGES; d = s; t = 0; }
  const float4* xs = (const float4*)(xsrc + (size_t)s * HID + h * CHD);
  const float4* xd = (const float4*)(xdst + (size_t)d * HID + h * CHD);
  const float4* as4 = (const float4*)(s_as + t * HID + h * CHD);
  const float4* ad4 = (const float4*)(s_ad + t * HID + h * CHD);
  float acc = 0.f;
#pragma unroll
  for (int j = 0; j < 8; ++j) {
    float4 a = xs[j], b = as4[j];
    acc += a.x * b.x + a.y * b.y + a.z * b.z + a.w * b.w;
    float4 c = xd[j], dd = ad4[j];
    acc += c.x * dd.x + c.y * dd.y + c.z * dd.z + c.w * dd.w;
  }
  float alpha = (acc >= 0.f) ? acc : 0.2f * acc;    // leaky_relu(0.2)
  alpha_out[(size_t)e * NHEAD + h] = alpha;
  atomicMax(&menc[(size_t)d * NHEAD + h], enc_f(alpha));
}

// ---------------------------------------------------------------------------
// Kernel 4: a = exp(alpha - max); denom += a
// ---------------------------------------------------------------------------
__global__ __launch_bounds__(256) void edge_exp_kernel(
    const int* __restrict__ dst, float* __restrict__ alpha,
    const unsigned* __restrict__ menc, float* __restrict__ denom) {
  const int gid = blockIdx.x * 256 + threadIdx.x;
  const int e = gid >> 3, h = gid & 7;
  if (e >= ET_EDGES) return;
  const int d = (e < E_EDGES) ? dst[e] : (e - E_EDGES);
  const float m = dec_f(menc[(size_t)d * NHEAD + h]);
  const float a = expf(alpha[(size_t)e * NHEAD + h] - m);
  alpha[(size_t)e * NHEAD + h] = a;
  atomicAdd(&denom[(size_t)d * NHEAD + h], a);
}

// ---------------------------------------------------------------------------
// Kernel 5: normalize, weight, scatter messages (one wave per edge)
// ---------------------------------------------------------------------------
__global__ __launch_bounds__(256) void edge_scatter_kernel(
    const int* __restrict__ src, const int* __restrict__ dst,
    const float* __restrict__ ew, const float* __restrict__ xsrc,
    const float* __restrict__ alpha, const float* __restrict__ denom,
    float* __restrict__ agg) {
  const int wid = (blockIdx.x * 256 + threadIdx.x) >> 5;
  const int lane = threadIdx.x & 31;
  if (wid >= ET_EDGES) return;
  int s, d; float w;
  if (wid < E_EDGES) { s = src[wid]; d = dst[wid]; w = ew[wid]; }
  else               { s = wid - E_EDGES; d = s; w = 1.f; }
  const float* xs = xsrc + (size_t)s * HID;
  float* outrow = agg + (size_t)d * HID;
#pragma unroll
  for (int h = 0; h < NHEAD; ++h) {
    float a = alpha[(size_t)wid * NHEAD + h] / denom[(size_t)d * NHEAD + h] * w;
    atomicAdd(&outrow[h * CHD + lane], xs[h * CHD + lane] * a);
  }
}

// ---------------------------------------------------------------------------
// Kernel 6: h = LN1(agg + bias + x)   (one wave per node)
// ---------------------------------------------------------------------------
__global__ __launch_bounds__(256) void ln1_kernel(
    const float* __restrict__ agg, const float* __restrict__ x,
    const float* __restrict__ bias, const float* __restrict__ g,
    const float* __restrict__ b, float* __restrict__ hout) {
  const int wid = (blockIdx.x * 256 + threadIdx.x) >> 5;
  const int lane = threadIdx.x & 31;
  if (wid >= N_NODES) return;
  const float* ar = agg + (size_t)wid * HID;
  const float* xr = x + (size_t)wid * HID;
  float v[8]; float s = 0.f;
#pragma unroll
  for (int j = 0; j < 8; ++j) {
    int c = lane + j * 32;
    v[j] = ar[c] + bias[c] + xr[c];
    s += v[j];
  }
  const float mu = wave_sum(s) * (1.f / 256.f);
  float vs = 0.f;
#pragma unroll
  for (int j = 0; j < 8; ++j) { float dl = v[j] - mu; vs += dl * dl; }
  const float rs = rsqrtf(wave_sum(vs) * (1.f / 256.f) + 1e-5f);
#pragma unroll
  for (int j = 0; j < 8; ++j) {
    int c = lane + j * 32;
    hout[(size_t)wid * HID + c] = (v[j] - mu) * rs * g[c] + b[c];
  }
}

// ---------------------------------------------------------------------------
// Kernel 7: fused FFN (256->1024 gelu ->256) + residual + LN2, 16 nodes/block.
// h tile staged via TDM (tensor_load_to_lds); [N,1024] intermediate stays in LDS.
// ---------------------------------------------------------------------------
__global__ __launch_bounds__(256) void ffn_ln2_kernel(
    const float* __restrict__ h,
    const __bf16* __restrict__ w1h, const __bf16* __restrict__ w1l,
    const float* __restrict__ b1,
    const __bf16* __restrict__ w2h, const __bf16* __restrict__ w2l,
    const float* __restrict__ b2, const float* __restrict__ g2,
    const float* __restrict__ beta2, float* __restrict__ out) {
  __shared__ float  lds_h[16 * HSTR];
  __shared__ __bf16 lds_ah[16 * AHSTR], lds_al[16 * AHSTR];
  __shared__ __bf16 lds_gh[16 * GSTR],  lds_gl[16 * GSTR];
  const int tid = threadIdx.x;
  const int m0 = blockIdx.x * 16;
  const float* htile = h + (size_t)m0 * HID;

  // --- stage fp32 h tile (16x256, contiguous 16 KB) into LDS via TDM ---
#if __has_builtin(__builtin_amdgcn_tensor_load_to_lds)
  if (tid < 32) {  // wave 0 issues the DMA and waits on TENSORcnt
    const unsigned lds_addr = (unsigned)(uintptr_t)(const void*)&lds_h[0];
    const unsigned long long ga = (unsigned long long)(uintptr_t)htile;
    // D# group0: count=1 | lds_addr | global_addr[56:0] | type=2
    u32x4 g0 = {1u, lds_addr, (unsigned)ga, (unsigned)(ga >> 32) | 0x80000000u};
    // D# group1: data_size=4B; tensor_dim0=256, tensor_dim1=16;
    //            tile_dim0=256, tile_dim1=16; tensor_dim0_stride=256
    i32x8 g1 = {0x00020000, 0x01000000, 0x00100000, 0x01000000, 16, 256, 0, 0};
    i32x4 gz4 = {0, 0, 0, 0};
    i32x8 gz8 = {0, 0, 0, 0, 0, 0, 0, 0};
    __builtin_amdgcn_tensor_load_to_lds(g0, g1, gz4, gz4, gz8, 0);
    __builtin_amdgcn_s_wait_tensorcnt(0);
  }
#else
  for (int i = tid; i < 16 * 64; i += 256)
    ((float4*)lds_h)[i] = ((const float4*)htile)[i];
#endif
  __syncthreads();

  // --- split the fp32 tile into bf16 hi/lo A-fragment arrays ---
  {
    const int row = tid >> 4, colb = (tid & 15) * 16;
    const float* s = lds_h + row * HSTR + colb;
    float f[16];
#pragma unroll
    for (int j = 0; j < 16; ++j) f[j] = s[j];
    v8bf h0, h1, l0, l1;
    split8(f, h0, l0); split8(f + 8, h1, l1);
    *(v8bf*)(lds_ah + row * AHSTR + colb) = h0;
    *(v8bf*)(lds_ah + row * AHSTR + colb + 8) = h1;
    *(v8bf*)(lds_al + row * AHSTR + colb) = l0;
    *(v8bf*)(lds_al + row * AHSTR + colb + 8) = l1;
  }
  __syncthreads();

  const int w = tid >> 5, lane = tid & 31, nl = lane & 15;
  const int co = w * 32;                              // this wave's 32 output cols
  const __bf16* ahrow = lds_ah + nl * AHSTR;
  const __bf16* alrow = lds_al + nl * AHSTR;
  const __bf16* ghrow = lds_gh + nl * GSTR;
  const __bf16* glrow = lds_gl + nl * GSTR;
  const int kba = (lane & 16) ? 16 : 0;
  const int kbb = (lane & 16) ? 16 : 0;
  v8f c0 = {0.f, 0.f, 0.f, 0.f, 0.f, 0.f, 0.f, 0.f};
  v8f c1 = {0.f, 0.f, 0.f, 0.f, 0.f, 0.f, 0.f, 0.f};

  for (int f0 = 0; f0 < FFN_DIM; f0 += 128) {
    // --- GEMM1: 16x16 tile of gelu(h @ W1^T + b1), cols f0 + w*16 + nl ---
    const int fcol = f0 + w * 16 + nl;
    const __bf16* w1hr = w1h + (size_t)fcol * 256;
    const __bf16* w1lr = w1l + (size_t)fcol * 256;
    if (f0 + 128 < FFN_DIM)
      __builtin_prefetch(w1h + (size_t)(fcol + 128) * 256, 0, 1);
    v8f gacc = {0.f, 0.f, 0.f, 0.f, 0.f, 0.f, 0.f, 0.f};
    for (int k0 = 0; k0 < 256; k0 += 32) {
      v16bf ah = load_a16(ahrow, k0 + kba);
      v16bf al = load_a16(alrow, k0 + kba);
      v16bf bh = load_b16(w1hr, k0 + kbb);
      v16bf bl = load_b16(w1lr, k0 + kbb);
      gacc = wmma3(ah, al, bh, bl, gacc);
    }
    __syncthreads();   // previous chunk's lds_g fully consumed
    {
      const int gr = (lane & 16) ? 8 : 0;
      const float bb = b1[fcol];
#pragma unroll
      for (int r = 0; r < 8; ++r) {
        float xg = gacc[r] + bb;
        float ge = 0.5f * xg * (1.f + erff(xg * 0.70710678118654752f));  // exact gelu
        __bf16 gh = (__bf16)ge;
        lds_gh[(r + gr) * GSTR + w * 16 + nl] = gh;
        lds_gl[(r + gr) * GSTR + w * 16 + nl] = (__bf16)(ge - (float)gh);
      }
    }
    __syncthreads();
    // --- GEMM2: accumulate this K-chunk into out cols co..co+31 ---
    const __bf16* w2hr0 = w2h + (size_t)(co + nl) * FFN_DIM + f0;
    const __bf16* w2lr0 = w2l + (size_t)(co + nl) * FFN_DIM + f0;
    const __bf16* w2hr1 = w2h + (size_t)(co + 16 + nl) * FFN_DIM + f0;
    const __bf16* w2lr1 = w2l + (size_t)(co + 16 + nl) * FFN_DIM + f0;
    for (int k0 = 0; k0 < 128; k0 += 32) {
      v16bf ah = load_a16(ghrow, k0 + kba);
      v16bf al = load_a16(glrow, k0 + kba);
      c0 = wmma3(ah, al, load_b16(w2hr0, k0 + kbb), load_b16(w2lr0, k0 + kbb), c0);
      c1 = wmma3(ah, al, load_b16(w2hr1, k0 + kbb), load_b16(w2lr1, k0 + kbb), c1);
    }
  }

  // residual: lds_h += ffn + b2   (each (row,col) touched by exactly one lane)
  {
    const int rbase = (lane & 16) ? 8 : 0;
    const float bb0 = b2[co + nl], bb1 = b2[co + 16 + nl];
#pragma unroll
    for (int r = 0; r < 8; ++r) {
      lds_h[(rbase + r) * HSTR + co + nl] += c0[r] + bb0;
      lds_h[(rbase + r) * HSTR + co + 16 + nl] += c1[r] + bb1;
    }
  }
  __syncthreads();

  // LN2 on the 16 rows (2 rows per wave), write final output
#pragma unroll
  for (int rr = 0; rr < 2; ++rr) {
    const int row = w * 2 + rr;
    const float* rp = lds_h + row * HSTR;
    float v[8]; float s = 0.f;
#pragma unroll
    for (int j = 0; j < 8; ++j) { v[j] = rp[lane + j * 32]; s += v[j]; }
    const float mu = wave_sum(s) * (1.f / 256.f);
    float vs = 0.f;
#pragma unroll
    for (int j = 0; j < 8; ++j) { float dl = v[j] - mu; vs += dl * dl; }
    const float rs = rsqrtf(wave_sum(vs) * (1.f / 256.f) + 1e-5f);
#pragma unroll
    for (int j = 0; j < 8; ++j) {
      int c = lane + j * 32;
      out[(size_t)(m0 + row) * HID + c] = (v[j] - mu) * rs * g2[c] + beta2[c];
    }
  }
}

// ---------------------------------------------------------------------------
extern "C" void kernel_launch(void* const* d_in, const int* in_sizes, int n_in,
                              void* d_out, int out_size, void* d_ws, size_t ws_size,
                              hipStream_t stream) {
  (void)in_sizes; (void)n_in; (void)out_size; (void)ws_size;
  const float* x     = (const float*)d_in[0];
  const int*   eidx  = (const int*)d_in[1];
  const int*   etype = (const int*)d_in[2];
  const float* ew    = (const float*)d_in[3];
  const float* Wsrc  = (const float*)d_in[4];
  const float* Wdst  = (const float*)d_in[5];
  const float* att_s = (const float*)d_in[6];
  const float* att_d = (const float*)d_in[7];
  const float* bias  = (const float*)d_in[8];
  const float* ln1g  = (const float*)d_in[9];
  const float* ln1b  = (const float*)d_in[10];
  const float* ln2g  = (const float*)d_in[11];
  const float* ln2b  = (const float*)d_in[12];
  const float* W1    = (const float*)d_in[13];
  const float* b1    = (const float*)d_in[14];
  const float* W2    = (const float*)d_in[15];
  const float* b2    = (const float*)d_in[16];
  const int* src = eidx;
  const int* dst = eidx + E_EDGES;

  // workspace carve (~187 MB)
  char* p = (char*)d_ws;
  float*    xsrc  = (float*)p;    p += (size_t)N_NODES * HID * sizeof(float);
  float*    xdst  = (float*)p;    p += (size_t)N_NODES * HID * sizeof(float);
  float*    alpha = (float*)p;    p += (size_t)ET_EDGES * NHEAD * sizeof(float);
  unsigned* menc  = (unsigned*)p; p += (size_t)N_NODES * NHEAD * sizeof(unsigned);
  float*    denom = (float*)p;    p += (size_t)N_NODES * NHEAD * sizeof(float);
  float*    hbuf  = (float*)p;    p += (size_t)N_NODES * HID * sizeof(float);
  __bf16* wsrch = (__bf16*)p;     p += (size_t)HID * HID * sizeof(__bf16);
  __bf16* wsrcl = (__bf16*)p;     p += (size_t)HID * HID * sizeof(__bf16);
  __bf16* wdsth = (__bf16*)p;     p += (size_t)HID * HID * sizeof(__bf16);
  __bf16* wdstl = (__bf16*)p;     p += (size_t)HID * HID * sizeof(__bf16);
  __bf16* w1h   = (__bf16*)p;     p += (size_t)FFN_DIM * HID * sizeof(__bf16);
  __bf16* w1l   = (__bf16*)p;     p += (size_t)FFN_DIM * HID * sizeof(__bf16);
  __bf16* w2h   = (__bf16*)p;     p += (size_t)HID * FFN_DIM * sizeof(__bf16);
  __bf16* w2l   = (__bf16*)p;     p += (size_t)HID * FFN_DIM * sizeof(__bf16);
  float* agg = (float*)d_out;     // reuse output buffer as message aggregator

  // 0) one-time weight splits
  cvt_weights_kernel<<<(HID * HID + 255) / 256, 256, 0, stream>>>(Wsrc, wsrch, wsrcl, HID * HID);
  cvt_weights_kernel<<<(HID * HID + 255) / 256, 256, 0, stream>>>(Wdst, wdsth, wdstl, HID * HID);
  cvt_weights_kernel<<<(FFN_DIM * HID + 255) / 256, 256, 0, stream>>>(W1, w1h, w1l, FFN_DIM * HID);
  cvt_weights_kernel<<<(FFN_DIM * HID + 255) / 256, 256, 0, stream>>>(W2, w2h, w2l, FFN_DIM * HID);
  // 1) projections (WMMA bf16-split GEMMs)
  gemm16_k256<<<dim3(N_NODES / 16, HID / 128), 256, 0, stream>>>(x, wsrch, wsrcl, xsrc, HID);
  gemm16_k256<<<dim3(N_NODES / 16, HID / 128), 256, 0, stream>>>(x, wdsth, wdstl, xdst, HID);
  // 2) init
  init_kernel<<<(N_NODES * HID + 255) / 256, 256, 0, stream>>>(agg, menc, denom);
  // 3-5) edge phase
  const int eh_blocks = (ET_EDGES * NHEAD + 255) / 256;
  edge_alpha_kernel<<<eh_blocks, 256, 0, stream>>>(src, dst, etype, xsrc, xdst,
                                                   att_s, att_d, alpha, menc);
  edge_exp_kernel<<<eh_blocks, 256, 0, stream>>>(dst, alpha, menc, denom);
  edge_scatter_kernel<<<(ET_EDGES + 7) / 8, 256, 0, stream>>>(src, dst, ew, xsrc,
                                                              alpha, denom, agg);
  // 6) LN1
  ln1_kernel<<<(N_NODES + 7) / 8, 256, 0, stream>>>(agg, x, bias, ln1g, ln1b, hbuf);
  // 7) fused FFN + residual + LN2 (WMMA + TDM), writes final output
  ffn_ln2_kernel<<<N_NODES / 16, 256, 0, stream>>>(hbuf, w1h, w1l, b1, w2h, w2l,
                                                   b2, ln2g, ln2b, (float*)d_out);
}
